// GCN_15977278341936
// MI455X (gfx1250) — matrix operand order repaired and checked by reading, not compiled, not run on hardware
//
#include <hip/hip_runtime.h>
#include <hip/hip_bf16.h>

// ---------------------------------------------------------------------------
// GraphSAGE (3-layer) inference for MI455X / gfx1250.
//  * Dense math: v_wmma_f32_16x16x32_bf16 (f32 accumulate).
//  * A-tiles staged to LDS via global_load_async_to_lds_b128 (+ s_wait_asynccnt);
//    fragments read back as paired ds_load_b128.
//  * Weights pre-packed into WMMA B-fragment register order -> each fragment
//    is one coalesced 32B load per lane (no packing movs).
//  * Aggregation AFTER the linear transform (mean commutes with linear):
//    conv2 scatters 64-wide rows; edge phase uses no-return f32 L2 atomics.
// ---------------------------------------------------------------------------

typedef __attribute__((ext_vector_type(16))) __bf16 v16bf;
typedef __attribute__((ext_vector_type(8)))  __bf16 v8bf;
typedef __attribute__((ext_vector_type(4)))  __bf16 v4bf;
typedef __attribute__((ext_vector_type(8)))  float  v8f;

__device__ __forceinline__ __bf16 to_bf16_rne(float f) {
    unsigned u = __float_as_uint(f);
    unsigned r = (u + 0x7FFFu + ((u >> 16) & 1u)) >> 16;
    unsigned short s = (unsigned short)r;
    return __builtin_bit_cast(__bf16, s);
}

// ------------------------- tiny utility kernels ----------------------------

__global__ void sage_fill_zero(float* __restrict__ p, long long n) {
    long long i = (long long)blockIdx.x * blockDim.x + threadIdx.x;
    if (i < n) p[i] = 0.0f;
}

// Pack W [128, NOUT] f32 row-major into WMMA B-fragment order (bf16):
//   Wp[((ct*4 + kk)*32 + lane)*16 + j] = W[(kk*32 + (lane&16?16:0) + j)*NOUT
//                                          + ct*16 + (lane&15)]
template <int NOUT>
__global__ void sage_w_pack(const float* __restrict__ W,
                            __bf16* __restrict__ Wp) {
    const int o = blockIdx.x * blockDim.x + threadIdx.x;
    if (o >= 128 * NOUT) return;
    const int j    = o & 15;
    const int lane = (o >> 4) & 31;
    const int kk   = (o >> 9) & 3;
    const int ct   = o >> 11;
    const int k    = kk * 32 + ((lane & 16) ? 16 : 0) + j;
    const int col  = ct * 16 + (lane & 15);
    Wp[o] = to_bf16_rne(W[k * NOUT + col]);
}

// features f32 -> bf16 (4 elements/thread), optional fused ReLU
template <bool RELU>
__global__ void sage_feat_bf16(const float* __restrict__ in,
                               __bf16* __restrict__ out, long long n4) {
    long long i = (long long)blockIdx.x * blockDim.x + threadIdx.x;
    if (i >= n4) return;
    const float4 v = *reinterpret_cast<const float4*>(in + i * 4);
    float a = v.x, b = v.y, c = v.z, d = v.w;
    if (RELU) {
        a = fmaxf(a, 0.f); b = fmaxf(b, 0.f);
        c = fmaxf(c, 0.f); d = fmaxf(d, 0.f);
    }
    v4bf o;
    o[0] = to_bf16_rne(a); o[1] = to_bf16_rne(b);
    o[2] = to_bf16_rne(c); o[3] = to_bf16_rne(d);
    *reinterpret_cast<v4bf*>(out + i * 4) = o;
}

__global__ void sage_degree(const int* __restrict__ dst,
                            float* __restrict__ deg, long long E) {
    long long e = (long long)blockIdx.x * blockDim.x + threadIdx.x;
    if (e < E) unsafeAtomicAdd(&deg[dst[e]], 1.0f);
}

__global__ void sage_invdeg(float* __restrict__ deg, long long N) {
    long long i = (long long)blockIdx.x * blockDim.x + threadIdx.x;
    if (i < N) deg[i] = 1.0f / fmaxf(deg[i], 1.0f);
}

// ------------------------- WMMA dual GEMM ----------------------------------
// Y_L = A @ W_L (+biasL if BL) ; Y_R = A @ W_R (+biasR if BR), when DUAL.
// A: [nrows,128] bf16 row-major.  WL/WR: packed fragment order (sage_w_pack).
// One wave per 16x16 output tile; blockDim = 32*(NOUT/16).
// A-tile (16x128 bf16 = 4KB) staged to LDS with async-to-LDS DMA.

template <int NOUT, bool DUAL, bool BL, bool BR>
__global__ void sage_gemm(const __bf16* __restrict__ A, int nrows,
                          const __bf16* __restrict__ WL,
                          const __bf16* __restrict__ WR,
                          const float* __restrict__ biasL,
                          const float* __restrict__ biasR,
                          float* __restrict__ YL,
                          float* __restrict__ YR) {
    constexpr int DK = 128;
    __shared__ __bf16 sA[16 * DK];           // 4KB A tile

    int m0 = blockIdx.x * 16;                // row tile (clamped; EXEC stays full)
    if (m0 + 16 > nrows) m0 = (nrows > 16) ? (nrows - 16) : 0;

    // ---- stage A tile to LDS: one b128 async DMA per 16 bytes ----
    {
        const unsigned ldsbase = (unsigned)(uintptr_t)(void*)sA;
        const unsigned long long gbase =
            (unsigned long long)(uintptr_t)(A + (size_t)m0 * DK);
        for (int t = threadIdx.x; t < 256; t += blockDim.x) {
            const unsigned off = (unsigned)t * 16u;
            asm volatile("global_load_async_to_lds_b128 %0, %1, %2"
                         :: "v"(ldsbase + off), "v"(off), "s"(gbase)
                         : "memory");
        }
        asm volatile("s_wait_asynccnt 0" ::: "memory");
        __syncthreads();
    }

    const int lane = threadIdx.x & 31;
    const int wave = threadIdx.x >> 5;       // == column tile ct
    const int ncol = wave * 16 + (lane & 15);
    const int kA   = (lane & 16) ? 16 : 0;   // A-fragment K sub-base

    // ---- B fragments: one aligned 32B load per (matrix, kk) ----
    const __bf16* wbL = WL + ((size_t)wave * 4) * 512 + (size_t)lane * 16;
    const __bf16* wbR = WR + ((size_t)wave * 4) * 512 + (size_t)lane * 16;
    v16bf bL[4], bR[4];
#pragma unroll
    for (int kk = 0; kk < 4; ++kk) {
        bL[kk] = *reinterpret_cast<const v16bf*>(wbL + kk * 512);
        if (DUAL) bR[kk] = *reinterpret_cast<const v16bf*>(wbR + kk * 512);
    }

    // ---- k-loop: 2 LDS b128 reads + 1-2 WMMA per step ----
    v8f accL = {0.f, 0.f, 0.f, 0.f, 0.f, 0.f, 0.f, 0.f};
    v8f accR = {0.f, 0.f, 0.f, 0.f, 0.f, 0.f, 0.f, 0.f};
    const v8bf* sArow =
        reinterpret_cast<const v8bf*>(sA + (lane & 15) * DK);
#pragma unroll
    for (int kk = 0; kk < 4; ++kk) {
        const int kb = kk * 32 + kA;          // multiple of 8
        const v8bf lo = sArow[kb >> 3];
        const v8bf hi = sArow[(kb >> 3) + 1];
        const v16bf af = __builtin_shufflevector(
            lo, hi, 0, 1, 2, 3, 4, 5, 6, 7, 8, 9, 10, 11, 12, 13, 14, 15);
        accL = __builtin_amdgcn_wmma_f32_16x16x32_bf16(
            false, af, false, bL[kk], (short)0, accL, false, false);
        if (DUAL)
            accR = __builtin_amdgcn_wmma_f32_16x16x32_bf16(
                false, af, false, bR[kk], (short)0, accR, false, false);
    }

    // ---- epilogue: row = m0 + r + (lane&16 ? 8 : 0), col = ncol ----
    const int rbase = m0 + ((lane & 16) ? 8 : 0);
    const float addL = BL ? biasL[ncol] : 0.f;
    const float addR = (DUAL && BR) ? biasR[ncol] : 0.f;
#pragma unroll
    for (int r = 0; r < 8; ++r) {
        const long long row = rbase + r;
        YL[row * NOUT + ncol] = accL[r] + addL;
        if (DUAL) YR[row * NOUT + ncol] = accR[r] + addR;
    }
}

// ------------------------- edge scatter (sum) ------------------------------
// agg[dst[e]][:] += Y[src[e]][:]   (C channels, C/4 lanes/edge, float4 loads)

template <int C>
__global__ void sage_scatter(const int* __restrict__ src,
                             const int* __restrict__ dst,
                             const float* __restrict__ Y,
                             float* __restrict__ agg, long long E) {
    constexpr int TPE = C / 4;
    long long gid = (long long)blockIdx.x * blockDim.x + threadIdx.x;
    long long e = gid / TPE;
    if (e >= E) return;
    __builtin_prefetch(src + e + 2048, 0, 0);   // global_prefetch_b8
    const int c4 = (int)(gid % TPE) * 4;
    const int s = src[e];
    const int d = dst[e];
    const float4 v =
        *reinterpret_cast<const float4*>(Y + (long long)s * C + c4);
    float* b = agg + (long long)d * C + c4;
    unsafeAtomicAdd(b + 0, v.x);
    unsafeAtomicAdd(b + 1, v.y);
    unsafeAtomicAdd(b + 2, v.z);
    unsafeAtomicAdd(b + 3, v.w);
}

// ------------------------- combine (hidden convs) --------------------------
// out = relu(agg*invdeg + yr) + 0.2*inp     (C = 128, float4-vectorized)

__global__ void sage_combine(const float* __restrict__ agg,
                             const float* __restrict__ yr,
                             const float* __restrict__ invdeg,
                             const float* __restrict__ inp,
                             float* __restrict__ out, long long total4) {
    long long i = (long long)blockIdx.x * blockDim.x + threadIdx.x;
    if (i >= total4) return;
    const long long node = i >> 5;              // (i*4)/128
    const float id = invdeg[node];
    const float4 a = *reinterpret_cast<const float4*>(agg + i * 4);
    const float4 r = *reinterpret_cast<const float4*>(yr  + i * 4);
    const float4 p = *reinterpret_cast<const float4*>(inp + i * 4);
    float4 o;
    o.x = fmaxf(a.x * id + r.x, 0.f) + 0.2f * p.x;
    o.y = fmaxf(a.y * id + r.y, 0.f) + 0.2f * p.y;
    o.z = fmaxf(a.z * id + r.z, 0.f) + 0.2f * p.z;
    o.w = fmaxf(a.w * id + r.w, 0.f) + 0.2f * p.w;
    *reinterpret_cast<float4*>(out + i * 4) = o;
}

// ------------------------- final log-softmax (DOUT=64) ---------------------

__global__ void sage_logsoftmax(const float* __restrict__ agg,
                                const float* __restrict__ yr,
                                const float* __restrict__ invdeg,
                                float* __restrict__ out, int nrows) {
    const int lane = threadIdx.x & 31;
    const int wave = threadIdx.x >> 5;
    const int row = blockIdx.x * (blockDim.x >> 5) + wave;
    if (row >= nrows) return;
    const float id = invdeg[row];
    const long long base = (long long)row * 64;
    float v0 = agg[base + lane] * id + yr[base + lane];
    float v1 = agg[base + 32 + lane] * id + yr[base + 32 + lane];
    float m = fmaxf(v0, v1);
#pragma unroll
    for (int off = 16; off > 0; off >>= 1)
        m = fmaxf(m, __shfl_xor(m, off, 32));
    float s = expf(v0 - m) + expf(v1 - m);
#pragma unroll
    for (int off = 16; off > 0; off >>= 1)
        s += __shfl_xor(s, off, 32);
    const float ls = logf(s);
    out[base + lane]      = v0 - m - ls;
    out[base + 32 + lane] = v1 - m - ls;
}

// ---------------------------------------------------------------------------

static inline long long cdiv_ll(long long a, long long b) {
    return (a + b - 1) / b;
}

extern "C" void kernel_launch(void* const* d_in, const int* in_sizes, int n_in,
                              void* d_out, int out_size, void* d_ws,
                              size_t ws_size, hipStream_t stream) {
    (void)n_in; (void)out_size; (void)ws_size;

    const float* x    = (const float*)d_in[0];
    const int*   ei   = (const int*)d_in[1];   // edge_index staged as int32
    const float* w_in = (const float*)d_in[2];
    const float* b_in = (const float*)d_in[3];
    const float* w_l0 = (const float*)d_in[4];
    const float* w_r0 = (const float*)d_in[5];
    const float* b0   = (const float*)d_in[6];
    const float* w_l1 = (const float*)d_in[7];
    const float* w_r1 = (const float*)d_in[8];
    const float* b1   = (const float*)d_in[9];
    const float* w_l2 = (const float*)d_in[10];
    const float* w_r2 = (const float*)d_in[11];
    const float* b2   = (const float*)d_in[12];

    const int DH = 128, DOUT = 64;
    const long long N = in_sizes[0] / 128;
    const long long E = in_sizes[1] / 2;
    const int* src = ei;
    const int* dst = ei + E;

    // ---- workspace carve-up ----
    float* inp    = (float*)d_ws;                 // h_pre (residual source)
    float* feat   = inp  + (size_t)N * DH;        // h1 / h2
    float* yl     = feat + (size_t)N * DH;
    float* yr     = yl   + (size_t)N * DH;
    float* agg    = yr   + (size_t)N * DH;
    float* invdeg = agg  + (size_t)N * DH;        // N floats
    __bf16* ab    = (__bf16*)(invdeg + N);        // bf16 A matrix [N,128]
    __bf16* winB  = ab   + (size_t)N * DH;        // packed weights (bf16)
    __bf16* wl0B  = winB + 128 * 128;
    __bf16* wr0B  = wl0B + 128 * 128;
    __bf16* wl1B  = wr0B + 128 * 128;
    __bf16* wr1B  = wl1B + 128 * 128;
    __bf16* wl2B  = wr1B + 128 * 128;             // 128x64 packed
    __bf16* wr2B  = wl2B + 128 * 64;

    const int T = 256;
    const long long n4 = N * DH / 4;              // feature quads

    // ---- weight repack f32 -> bf16 fragment order ----
    sage_w_pack<128><<<cdiv_ll(128 * 128, T), T, 0, stream>>>(w_in, winB);
    sage_w_pack<128><<<cdiv_ll(128 * 128, T), T, 0, stream>>>(w_l0, wl0B);
    sage_w_pack<128><<<cdiv_ll(128 * 128, T), T, 0, stream>>>(w_r0, wr0B);
    sage_w_pack<128><<<cdiv_ll(128 * 128, T), T, 0, stream>>>(w_l1, wl1B);
    sage_w_pack<128><<<cdiv_ll(128 * 128, T), T, 0, stream>>>(w_r1, wr1B);
    sage_w_pack<64><<<cdiv_ll(128 * 64, T), T, 0, stream>>>(w_l2, wl2B);
    sage_w_pack<64><<<cdiv_ll(128 * 64, T), T, 0, stream>>>(w_r2, wr2B);

    // ---- degree (shared by all three convs) ----
    sage_fill_zero<<<cdiv_ll(N, T), T, 0, stream>>>(invdeg, N);
    sage_degree<<<cdiv_ll(E, T), T, 0, stream>>>(dst, invdeg, E);
    sage_invdeg<<<cdiv_ll(N, T), T, 0, stream>>>(invdeg, N);

    const int rowTiles = (int)cdiv_ll(N, 16);

    // ---- input linear: inp = x @ w_in + b_in ----
    sage_feat_bf16<false><<<cdiv_ll(n4, T), T, 0, stream>>>(x, ab, n4);
    sage_gemm<128, false, true, false><<<rowTiles, 256, 0, stream>>>(
        ab, (int)N, winB, winB, b_in, b_in, inp, inp);

    // ---- conv0: A = relu(inp) ----
    sage_feat_bf16<true><<<cdiv_ll(n4, T), T, 0, stream>>>(inp, ab, n4);
    sage_gemm<128, true, false, true><<<rowTiles, 256, 0, stream>>>(
        ab, (int)N, wl0B, wr0B, b0, b0, yl, yr);
    sage_fill_zero<<<cdiv_ll(N * DH, T), T, 0, stream>>>(agg, N * DH);
    sage_scatter<128><<<cdiv_ll(E * 32, T), T, 0, stream>>>(src, dst, yl, agg, E);
    sage_combine<<<cdiv_ll(n4, T), T, 0, stream>>>(agg, yr, invdeg, inp, feat, n4);

    // ---- conv1: A = feat ----
    sage_feat_bf16<false><<<cdiv_ll(n4, T), T, 0, stream>>>(feat, ab, n4);
    sage_gemm<128, true, false, true><<<rowTiles, 256, 0, stream>>>(
        ab, (int)N, wl1B, wr1B, b1, b1, yl, yr);
    sage_fill_zero<<<cdiv_ll(N * DH, T), T, 0, stream>>>(agg, N * DH);
    sage_scatter<128><<<cdiv_ll(E * 32, T), T, 0, stream>>>(src, dst, yl, agg, E);
    sage_combine<<<cdiv_ll(n4, T), T, 0, stream>>>(agg, yr, invdeg, inp, feat, n4);

    // ---- conv2 (128 -> 64): transform first, aggregate 64-wide ----
    sage_feat_bf16<false><<<cdiv_ll(n4, T), T, 0, stream>>>(feat, ab, n4);
    sage_gemm<64, true, false, true><<<rowTiles, 128, 0, stream>>>(
        ab, (int)N, wl2B, wr2B, b2, b2, yl, yr);
    sage_fill_zero<<<cdiv_ll(N * DOUT, T), T, 0, stream>>>(agg, N * DOUT);
    sage_scatter<64><<<cdiv_ll(E * 16, T), T, 0, stream>>>(src, dst, yl, agg, E);

    // ---- logits + log_softmax -> d_out ----
    sage_logsoftmax<<<cdiv_ll(N, 8), 256, 0, stream>>>(agg, yr, invdeg,
                                                       (float*)d_out, (int)N);
}
